// DynamicsSolver_82429012345241
// MI455X (gfx1250) — matrix operand order repaired and checked by reading, not compiled, not run on hardware
//
#include <hip/hip_runtime.h>
#include <hip/hip_bf16.h>

typedef __attribute__((ext_vector_type(16))) _Float16 v16h;
typedef __attribute__((ext_vector_type(8)))  float    v8f;

static constexpr int NN = 20000;
static constexpr int EE = 320000;

// ---------------------------------------------------------------- small vec3
struct V3 { float x, y, z; };
__device__ inline V3 operator+(V3 a, V3 b){ return {a.x+b.x, a.y+b.y, a.z+b.z}; }
__device__ inline V3 operator-(V3 a, V3 b){ return {a.x-b.x, a.y-b.y, a.z-b.z}; }
__device__ inline V3 operator*(V3 a, float s){ return {a.x*s, a.y*s, a.z*s}; }
__device__ inline float dot3(V3 a, V3 b){ return a.x*b.x + a.y*b.y + a.z*b.z; }
__device__ inline V3 cross3(V3 a, V3 b){
  return {a.y*b.z - a.z*b.y, a.z*b.x - a.x*b.z, a.x*b.y - a.y*b.x};
}
__device__ inline V3 nrm3(V3 v){
  float l = sqrtf(dot3(v, v));
  return v * (1.f / fmaxf(l, 1e-6f));
}
__device__ inline V3 ld3(const float* p, int i){ return {p[i*3], p[i*3+1], p[i*3+2]}; }

// ------------------------------------------------- WMMA tile layer (16 rows)
// X: f16 LDS, row-major, stride xs halfs; Kp = NK*32 (zero padded).
// Wp: packed B fragments: frag(nt,kc) = Wp + ((nt*NK+kc)*32 + lane)*16 halfs,
//     lane&15 = output column within ntile, lanes<16 hold K 0..7,16..23, >=16 hold 8..15,24..31.
// Y: f32 LDS 16x128 row-major; bias added per column.
template<int NK>
__device__ inline void mlp_layer(const _Float16* __restrict__ X, int xs,
                                 const _Float16* __restrict__ Wp,
                                 const float* __restrict__ bias,
                                 int lane, float* __restrict__ Y) {
  const int row = lane & 15;
  const int kb8 = (lane & 16) ? 8 : 0;   // A-fragment K base for hi half-wave
  const int rb  = (lane & 16) ? 8 : 0;   // D-fragment row base for hi half-wave
  #pragma unroll
  for (int nt = 0; nt < 8; ++nt) {
    v8f acc = {};
    #pragma unroll
    for (int kc = 0; kc < NK; ++kc) {
      const _Float16* ap = X + row * xs + kc * 32 + kb8;
      v16h a;
      __builtin_memcpy(&a, ap, 16);                       // K = kb..kb+7
      __builtin_memcpy((char*)&a + 16, ap + 16, 16);      // K = kb+16..kb+23
      const _Float16* bp = Wp + ((size_t)(nt * NK + kc) * 32 + lane) * 16;
      v16h b;
      __builtin_memcpy(&b, bp, 32);
      acc = __builtin_amdgcn_wmma_f32_16x16x32_f16(
          false, a, false, b, (short)0, acc, false, false);
    }
    const int col = nt * 16 + (lane & 15);
    const float bv = bias[col];
    #pragma unroll
    for (int g = 0; g < 8; ++g) Y[(rb + g) * 128 + col] = acc[g] + bv;
  }
}

// Single-ntile head layer: 16x128 @ 128x16 (D cols used, rest zero-padded).
template<int NK>
__device__ inline void head_layer(const _Float16* __restrict__ X, int xs,
                                  const _Float16* __restrict__ Wp,
                                  int lane, float* __restrict__ Y2 /*16x16*/) {
  const int row = lane & 15;
  const int kb8 = (lane & 16) ? 8 : 0;
  const int rb  = (lane & 16) ? 8 : 0;
  v8f acc = {};
  #pragma unroll
  for (int kc = 0; kc < NK; ++kc) {
    const _Float16* ap = X + row * xs + kc * 32 + kb8;
    v16h a;
    __builtin_memcpy(&a, ap, 16);
    __builtin_memcpy((char*)&a + 16, ap + 16, 16);
    const _Float16* bp = Wp + ((size_t)kc * 32 + lane) * 16;
    v16h b;
    __builtin_memcpy(&b, bp, 32);
    acc = __builtin_amdgcn_wmma_f32_16x16x32_f16(
        false, a, false, b, (short)0, acc, false, false);
  }
  const int col = lane & 15;
  #pragma unroll
  for (int g = 0; g < 8; ++g) Y2[(rb + g) * 16 + col] = acc[g];
}

__device__ inline void relu_to_f16(const float* __restrict__ Y,
                                   _Float16* __restrict__ H, int lane) {
  for (int i = lane; i < 16 * 128; i += 32)
    H[i] = (_Float16)fmaxf(Y[i], 0.f);
}

// LayerNorm over 128 cols/row using the full wave: lanes r and r+16 split row r.
__device__ inline void ln_store(const float* __restrict__ Y,
                                const float* __restrict__ g,
                                const float* __restrict__ bt,
                                _Float16* __restrict__ out, int os,
                                bool accum, int lane) {
  const int row = lane & 15;
  const int hf  = (lane >> 4) & 1;           // 0: cols 0..63, 1: cols 64..127
  const float* yr = Y + row * 128 + hf * 64;
  float s = 0.f, s2 = 0.f;
  #pragma unroll 4
  for (int c = 0; c < 64; ++c) { float v = yr[c]; s += v; s2 += v * v; }
  s  += __shfl_xor(s, 16, 32);
  s2 += __shfl_xor(s2, 16, 32);
  const float mu  = s * (1.f / 128.f);
  const float var = s2 * (1.f / 128.f) - mu * mu;
  const float inv = rsqrtf(var + 1e-5f);
  _Float16* orow = out + row * os + hf * 64;
  const float* gg = g + hf * 64;
  const float* bb = bt + hf * 64;
  for (int c = 0; c < 64; ++c) {
    float v = (yr[c] - mu) * inv * gg[c] + bb[c];
    orow[c] = accum ? (_Float16)((float)orow[c] + v) : (_Float16)v;
  }
}

// ------------------------------------------------------------- weight packer
// W fp32 (K x 128 row-major) -> f16 B fragments, K zero-padded to NKc*32.
__global__ void pack_weight_kernel(const float* __restrict__ W,
                                   _Float16* __restrict__ dst,
                                   int K, int NKc, int total) {
  int idx = blockIdx.x * blockDim.x + threadIdx.x;
  if (idx >= total) return;
  int j    = idx & 15;
  int lane = (idx >> 4) & 31;
  int kc   = (idx >> 9) % NKc;
  int nt   = (idx >> 9) / NKc;
  int n = nt * 16 + (lane & 15);
  int k = kc * 32 + ((lane & 16) ? 8 : 0) + (j & 7) + ((j & 8) ? 16 : 0);
  dst[idx] = (k < K) ? (_Float16)W[k * 128 + n] : (_Float16)0.f;
}

// Head W2 fp32 (128 x D row-major) -> single-ntile fragments, cols >= D zeroed.
__global__ void pack_head_kernel(const float* __restrict__ W,
                                 _Float16* __restrict__ dst, int D) {
  int idx = blockIdx.x * blockDim.x + threadIdx.x;      // 4*512 = 2048 total
  if (idx >= 2048) return;
  int j    = idx & 15;
  int lane = (idx >> 4) & 31;
  int kc   = idx >> 9;                                  // 0..3
  int n = lane & 15;
  int k = kc * 32 + ((lane & 16) ? 8 : 0) + (j & 7) + ((j & 8) ? 16 : 0);
  dst[idx] = (n < D) ? (_Float16)W[k * D + n] : (_Float16)0.f;
}

// ------------------------------------------------------------- node pipeline
struct NodeArgs {
  const float* nfeat;
  const _Float16 *w1, *w2;                   // encoder packs
  const float *b1, *b2, *g, *bt;             // encoder bias / LN
  const _Float16 *wh[5];                     // head W1 packs: nw,invm,invi,extdv,vscale
  const float *bh[5];                        // head b1
  const _Float16 *w2p[5];                    // head W2 packs (128->16 padded)
  const float *b2h[5];                       // head b2
  _Float16* lat;                             // f16 latent out [N,128]
  float *wnode, *invm, *invi, *extdv, *vsc;
};

__global__ __launch_bounds__(64) void node_kernel(NodeArgs A) {
  __shared__ __align__(32) _Float16 sX0[2][16 * 32];
  __shared__ __align__(32) _Float16 sH [2][16 * 128];
  __shared__ __align__(32) float    sY [2][16 * 128];
  __shared__ __align__(32) float    sY2[2][16 * 16];
  __shared__ __align__(32) _Float16 sL [2][16 * 128];
  const int lane = threadIdx.x & 31, wv = threadIdx.x >> 5;
  const int row0 = (blockIdx.x * 2 + wv) * 16;
  _Float16* X0 = sX0[wv]; _Float16* H = sH[wv]; float* Y = sY[wv];
  float* Y2 = sY2[wv]; _Float16* LAT = sL[wv];

  for (int i = lane; i < 16 * 32; i += 32) {
    int r = i >> 5, c = i & 31;
    X0[i] = (c < 9) ? (_Float16)A.nfeat[(row0 + r) * 9 + c] : (_Float16)0.f;
  }
  __syncthreads();
  mlp_layer<1>(X0, 32, A.w1, A.b1, lane, Y);
  __syncthreads();
  relu_to_f16(Y, H, lane);
  __syncthreads();
  mlp_layer<4>(H, 128, A.w2, A.b2, lane, Y);
  __syncthreads();
  {                                           // LN -> LAT + global (full wave)
    const int row = lane & 15;
    const int hf  = (lane >> 4) & 1;
    const float* yr = Y + row * 128 + hf * 64;
    float s = 0.f, s2 = 0.f;
    #pragma unroll 4
    for (int c = 0; c < 64; ++c) { float v = yr[c]; s += v; s2 += v * v; }
    s  += __shfl_xor(s, 16, 32);
    s2 += __shfl_xor(s2, 16, 32);
    const float mu  = s * (1.f / 128.f);
    const float var = s2 * (1.f / 128.f) - mu * mu;
    const float inv = rsqrtf(var + 1e-5f);
    const int n = row0 + row;
    _Float16* lrow = LAT + row * 128 + hf * 64;
    _Float16* grow = A.lat + (size_t)n * 128 + hf * 64;
    const float* gg = A.g + hf * 64;
    const float* bb = A.bt + hf * 64;
    for (int c = 0; c < 64; ++c) {
      _Float16 v = (_Float16)((yr[c] - mu) * inv * gg[c] + bb[c]);
      lrow[c] = v;
      grow[c] = v;
    }
  }
  __syncthreads();

  const int   Dh[5]   = {1, 1, 1, 3, 1};
  float* const out[5] = {A.wnode, A.invm, A.invi, A.extdv, A.vsc};
  #pragma unroll
  for (int h = 0; h < 5; ++h) {
    mlp_layer<4>(LAT, 128, A.wh[h], A.bh[h], lane, Y);
    __syncthreads();
    relu_to_f16(Y, H, lane);
    __syncthreads();
    head_layer<4>(H, 128, A.w2p[h], lane, Y2);
    __syncthreads();
    if (lane < 16) {
      const int n = row0 + lane;
      for (int d = 0; d < Dh[h]; ++d)
        out[h][n * Dh[h] + d] = Y2[lane * 16 + d] + A.b2h[h][d];
    }
    __syncthreads();
  }
}

// ------------------------------------------------------------- edge pipeline
struct EdgeArgs {
  const int* ei;
  const float *pos, *vt, *vtm1, *om, *eattr;
  const float *vs_s, *dmin_s, *dmax_s;
  const _Float16 *wnf1, *wnf2, *wed1, *wed2, *win1, *win2;  // encoder packs
  const float *bnf1, *bnf2, *gnf, *btnf;
  const float *bed1, *bed2, *ged, *bted;
  const float *bin1, *bin2, *gin, *btin;
  const _Float16 *wh[4];                     // i1,i2,fscale,comp W1 packs
  const float *bh[4];                        // head b1
  const _Float16 *w2p[4];                    // head W2 packs (128->16 padded)
  const float *b2h[4];                       // head b2
  const _Float16* lat;
  const float* wnode;
  float *net_f, *net_t, *net_c, *cnt;
};

__global__ __launch_bounds__(64) void edge_kernel(EdgeArgs A) {
  __shared__ __align__(32) float    sFE[2][16 * 16];
  __shared__ __align__(32) float    sGA[2][16 * 20];
  __shared__              int       sSR[2][16 * 2];
  __shared__ __align__(32) float    sWW[2][16 * 2];
  __shared__ __align__(32) _Float16 sX0[2][16 * 32];
  __shared__ __align__(32) _Float16 sH [2][16 * 128];
  __shared__ __align__(32) float    sY [2][16 * 128];
  __shared__ __align__(32) float    sY2[2][16 * 16];
  __shared__ __align__(32) _Float16 sXI[2][16 * 384];
  __shared__ __align__(32) float    sCF[2][16 * 12];

  const int lane = threadIdx.x & 31, wv = threadIdx.x >> 5;
  const int e0 = (blockIdx.x * 2 + wv) * 16;
  float* FE = sFE[wv]; float* GA = sGA[wv]; int* SR = sSR[wv]; float* WW = sWW[wv];
  _Float16* X0 = sX0[wv]; _Float16* H = sH[wv]; float* Y = sY[wv]; float* Y2 = sY2[wv];
  _Float16* XI = sXI[wv]; float* CF = sCF[wv];

  const float vs = A.vs_s[0], dmin = A.dmin_s[0], dmax = A.dmax_s[0];

  // ---- geometry + raw features (lanes 0..15, one edge each)
  if (lane < 16) {
    const int e = e0 + lane;
    const int si = A.ei[e], ri = A.ei[EE + e];
    V3 sp = ld3(A.pos, si),  rp = ld3(A.pos, ri);
    V3 sv = ld3(A.vt, si),   rv = ld3(A.vt, ri);
    V3 spv = ld3(A.vtm1, si), rpv = ld3(A.vtm1, ri);
    V3 sw = ld3(A.om, si),   rw = ld3(A.om, ri);
    V3 rel = rp - sp;
    float dist = sqrtf(dot3(rel, rel));
    V3 a = rel * (1.f / dist);
    V3 b = nrm3(cross3(rv - sv, a)) + nrm3(sv + rv)
         + nrm3(cross3(rw - sw, a)) + nrm3(sw + rw)
         + nrm3(cross3(rpv - spv, a)) + nrm3(spv + rpv);
    V3 bprl = a * dot3(b, a);
    V3 bprp = b - bprl;
    V3 vb = nrm3(cross3(bprp, a));
    V3 vc = nrm3(cross3(bprl, vb));
    V3 sv_ = sv * (1.f / vs), rv_ = rv * (1.f / vs);
    float* fe = FE + lane * 16;
    fe[0] = dot3(a, sv_); fe[1] = dot3(vb, sv_); fe[2] = dot3(vc, sv_);
    fe[3] = dot3(a, sw);  fe[4] = dot3(vb, sw);  fe[5] = dot3(vc, sw);
    fe[6] = -dot3(a, rv_); fe[7] = -dot3(vb, rv_); fe[8] = -dot3(vc, rv_);
    fe[9] = -dot3(a, rw);  fe[10] = -dot3(vb, rw); fe[11] = -dot3(vc, rw);
    float sm = (dist - dmin) / (dmax - dmin);
    fe[12] = fabsf(sm);
    fe[13] = A.eattr[e * 3]; fe[14] = A.eattr[e * 3 + 1]; fe[15] = A.eattr[e * 3 + 2];
    float* ga = GA + lane * 20;
    ga[0]=a.x; ga[1]=a.y; ga[2]=a.z; ga[3]=vb.x; ga[4]=vb.y; ga[5]=vb.z;
    ga[6]=vc.x; ga[7]=vc.y; ga[8]=vc.z;
    ga[9]=sp.x; ga[10]=sp.y; ga[11]=sp.z; ga[12]=rp.x; ga[13]=rp.y; ga[14]=rp.z;
    SR[lane * 2] = si; SR[lane * 2 + 1] = ri;
    WW[lane * 2] = A.wnode[si]; WW[lane * 2 + 1] = A.wnode[ri];
  }
  __syncthreads();

  // ---- node-latent gather: XI[:,128:256] = lat[s] + lat[r]
  for (int i = lane; i < 16 * 128; i += 32) {
    int r = i >> 7, c = i & 127;
    int si = SR[r * 2], ri = SR[r * 2 + 1];
    XI[r * 384 + 128 + c] =
        (_Float16)((float)A.lat[(size_t)si * 128 + c] + (float)A.lat[(size_t)ri * 128 + c]);
  }
  __syncthreads();

  // ---- nfeat MLP on s_feat -> XI[:,0:128]
  for (int i = lane; i < 16 * 32; i += 32) {
    int r = i >> 5, c = i & 31;
    X0[i] = (c < 6) ? (_Float16)FE[r * 16 + c] : (_Float16)0.f;
  }
  __syncthreads();
  mlp_layer<1>(X0, 32, A.wnf1, A.bnf1, lane, Y); __syncthreads();
  relu_to_f16(Y, H, lane); __syncthreads();
  mlp_layer<4>(H, 128, A.wnf2, A.bnf2, lane, Y); __syncthreads();
  ln_store(Y, A.gnf, A.btnf, XI, 384, false, lane); __syncthreads();

  // ---- nfeat MLP on r_feat -> XI[:,0:128] +=
  for (int i = lane; i < 16 * 32; i += 32) {
    int r = i >> 5, c = i & 31;
    X0[i] = (c < 6) ? (_Float16)FE[r * 16 + 6 + c] : (_Float16)0.f;
  }
  __syncthreads();
  mlp_layer<1>(X0, 32, A.wnf1, A.bnf1, lane, Y); __syncthreads();
  relu_to_f16(Y, H, lane); __syncthreads();
  mlp_layer<4>(H, 128, A.wnf2, A.bnf2, lane, Y); __syncthreads();
  ln_store(Y, A.gnf, A.btnf, XI, 384, true, lane); __syncthreads();

  // ---- edge MLP on e_feat -> XI[:,256:384]
  for (int i = lane; i < 16 * 32; i += 32) {
    int r = i >> 5, c = i & 31;
    X0[i] = (c < 4) ? (_Float16)FE[r * 16 + 12 + c] : (_Float16)0.f;
  }
  __syncthreads();
  mlp_layer<1>(X0, 32, A.wed1, A.bed1, lane, Y); __syncthreads();
  relu_to_f16(Y, H, lane); __syncthreads();
  mlp_layer<4>(H, 128, A.wed2, A.bed2, lane, Y); __syncthreads();
  ln_store(Y, A.ged, A.bted, XI + 256, 384, false, lane); __syncthreads();

  // ---- interaction MLP (384 -> 128 -> 128 + LN), result into XI[:,0:128]
  mlp_layer<12>(XI, 384, A.win1, A.bin1, lane, Y); __syncthreads();
  relu_to_f16(Y, H, lane); __syncthreads();
  mlp_layer<4>(H, 128, A.win2, A.bin2, lane, Y); __syncthreads();
  ln_store(Y, A.gin, A.btin, XI, 384, false, lane); __syncthreads();

  // ---- four decoder heads from inter (XI[:,0:128], stride 384)
  const int Dh[4]  = {3, 3, 1, 3};
  const int cfo[4] = {0, 3, 6, 7};
  #pragma unroll
  for (int h = 0; h < 4; ++h) {
    mlp_layer<4>(XI, 384, A.wh[h], A.bh[h], lane, Y);
    __syncthreads();
    relu_to_f16(Y, H, lane);
    __syncthreads();
    head_layer<4>(H, 128, A.w2p[h], lane, Y2);
    __syncthreads();
    if (lane < 16) {
      for (int d = 0; d < Dh[h]; ++d)
        CF[lane * 12 + cfo[h] + d] = Y2[lane * 16 + d] + A.b2h[h][d];
    }
    __syncthreads();
  }

  // ---- decode + scatter to receivers
  if (lane < 16) {
    const float* cf = CF + lane * 12;
    const float* ga = GA + lane * 20;
    V3 a  = {ga[0], ga[1], ga[2]};
    V3 vb = {ga[3], ga[4], ga[5]};
    V3 vc = {ga[6], ga[7], ga[8]};
    V3 sp = {ga[9], ga[10], ga[11]};
    V3 rp = {ga[12], ga[13], ga[14]};
    float ws_ = WW[lane * 2], wr_ = WW[lane * 2 + 1];
    V3 fij  = a * cf[0] + vb * cf[1] + vc * cf[2];
    V3 aij  = a * cf[3] + vb * cf[4] + vc * cf[5];
    float lam = cf[6];
    V3 dxij = a * cf[7] + vb * cf[8] + vc * cf[9];
    V3 r0 = (sp * ws_ + rp * wr_) * (1.f / (ws_ + wr_));
    V3 tau = aij - cross3(rp - r0, fij * lam);
    int ri = SR[lane * 2 + 1];
    unsafeAtomicAdd(&A.net_f[ri * 3 + 0], fij.x);
    unsafeAtomicAdd(&A.net_f[ri * 3 + 1], fij.y);
    unsafeAtomicAdd(&A.net_f[ri * 3 + 2], fij.z);
    unsafeAtomicAdd(&A.net_t[ri * 3 + 0], tau.x);
    unsafeAtomicAdd(&A.net_t[ri * 3 + 1], tau.y);
    unsafeAtomicAdd(&A.net_t[ri * 3 + 2], tau.z);
    unsafeAtomicAdd(&A.net_c[ri * 3 + 0], dxij.x);
    unsafeAtomicAdd(&A.net_c[ri * 3 + 1], dxij.y);
    unsafeAtomicAdd(&A.net_c[ri * 3 + 2], dxij.z);
    unsafeAtomicAdd(&A.cnt[ri], 1.f);
  }
}

// ------------------------------------------------------------------ finalize
__global__ void finalize_kernel(const float* __restrict__ vt, const float* __restrict__ vs_s,
                                const float* __restrict__ invm, const float* __restrict__ invi,
                                const float* __restrict__ extdv, const float* __restrict__ vsc,
                                const float* __restrict__ net_f, const float* __restrict__ net_t,
                                const float* __restrict__ net_c, const float* __restrict__ cnt,
                                float* __restrict__ out) {
  int n = blockIdx.x * blockDim.x + threadIdx.x;
  if (n >= NN) return;
  float vs = vs_s[0];
  float cn = fmaxf(cnt[n], 1.f);
  float im = invm[n], ii = invi[n], sc = vsc[n];
  #pragma unroll
  for (int d = 0; d < 3; ++d) {
    out[n * 3 + d]          = im * net_f[n * 3 + d];
    out[3 * NN + n * 3 + d] = ii * net_t[n * 3 + d];
    out[6 * NN + n * 3 + d] = (vt[n * 3 + d] / vs + extdv[n * 3 + d]) * sc
                            + net_c[n * 3 + d] / cn;
  }
}

// -------------------------------------------------------------------- launch
extern "C" void kernel_launch(void* const* d_in, const int* in_sizes, int n_in,
                              void* d_out, int out_size, void* d_ws, size_t ws_size,
                              hipStream_t stream) {
  (void)in_sizes; (void)n_in; (void)out_size; (void)ws_size;
  auto F = [&](int i) { return (const float*)d_in[i]; };
  const int* ei = (const int*)d_in[0];

  char* ws = (char*)d_ws;
  size_t off = 0;
  auto alloc = [&](size_t bytes) -> char* {
    char* p = ws + off;
    off = (off + bytes + 255) & ~(size_t)255;
    return p;
  };
  auto packed_bytes = [](int K) {
    int Kp = (K + 31) / 32 * 32;
    return (size_t)128 * Kp * 2;
  };

  // packed first-layer / encoder weights (B operands, 8 ntiles)
  _Float16* wp_n1  = (_Float16*)alloc(packed_bytes(9));
  _Float16* wp_n2  = (_Float16*)alloc(packed_bytes(128));
  _Float16* wp_nf1 = (_Float16*)alloc(packed_bytes(6));
  _Float16* wp_nf2 = (_Float16*)alloc(packed_bytes(128));
  _Float16* wp_e1  = (_Float16*)alloc(packed_bytes(4));
  _Float16* wp_e2  = (_Float16*)alloc(packed_bytes(128));
  _Float16* wp_in1 = (_Float16*)alloc(packed_bytes(384));
  _Float16* wp_in2 = (_Float16*)alloc(packed_bytes(128));
  _Float16* wp_i1  = (_Float16*)alloc(packed_bytes(128));
  _Float16* wp_i2  = (_Float16*)alloc(packed_bytes(128));
  _Float16* wp_fs  = (_Float16*)alloc(packed_bytes(128));
  _Float16* wp_cp  = (_Float16*)alloc(packed_bytes(128));
  _Float16* wp_nw  = (_Float16*)alloc(packed_bytes(128));
  _Float16* wp_im  = (_Float16*)alloc(packed_bytes(128));
  _Float16* wp_ii  = (_Float16*)alloc(packed_bytes(128));
  _Float16* wp_ex  = (_Float16*)alloc(packed_bytes(128));
  _Float16* wp_vs  = (_Float16*)alloc(packed_bytes(128));

  // packed head second layers (single ntile, 128->16 padded) : 2048 halfs each
  const size_t hb = (size_t)2048 * 2;
  _Float16* hp_i1 = (_Float16*)alloc(hb);
  _Float16* hp_i2 = (_Float16*)alloc(hb);
  _Float16* hp_fs = (_Float16*)alloc(hb);
  _Float16* hp_cp = (_Float16*)alloc(hb);
  _Float16* hp_nw = (_Float16*)alloc(hb);
  _Float16* hp_im = (_Float16*)alloc(hb);
  _Float16* hp_ii = (_Float16*)alloc(hb);
  _Float16* hp_ex = (_Float16*)alloc(hb);
  _Float16* hp_vs = (_Float16*)alloc(hb);

  _Float16* lat = (_Float16*)alloc((size_t)NN * 128 * 2);
  float* wnode = (float*)alloc((size_t)NN * 4);
  float* invm  = (float*)alloc((size_t)NN * 4);
  float* invi  = (float*)alloc((size_t)NN * 4);
  float* vsc   = (float*)alloc((size_t)NN * 4);
  float* extdv = (float*)alloc((size_t)NN * 3 * 4);
  float* net   = (float*)alloc((size_t)NN * 10 * 4);
  float* net_f = net;
  float* net_t = net + 3 * NN;
  float* net_c = net + 6 * NN;
  float* cnt   = net + 9 * NN;

  auto pack = [&](const float* W, _Float16* dst, int K) {
    int Kp = (K + 31) / 32 * 32, NKc = Kp / 32;
    int total = 8 * NKc * 512;
    pack_weight_kernel<<<(total + 255) / 256, 256, 0, stream>>>(W, dst, K, NKc, total);
  };
  auto packh = [&](const float* W, _Float16* dst, int D) {
    pack_head_kernel<<<8, 256, 0, stream>>>(W, dst, D);
  };
  pack(F(10), wp_n1, 9);    pack(F(12), wp_n2, 128);   // node_enc
  pack(F(22), wp_nf1, 6);   pack(F(24), wp_nf2, 128);  // nfeat_enc
  pack(F(16), wp_e1, 4);    pack(F(18), wp_e2, 128);   // edge_enc
  pack(F(28), wp_in1, 384); pack(F(30), wp_in2, 128);  // inter_enc
  pack(F(34), wp_i1, 128);  pack(F(38), wp_i2, 128);   // i1, i2 W1
  pack(F(42), wp_fs, 128);  pack(F(50), wp_cp, 128);   // fscale, comp W1
  pack(F(46), wp_nw, 128);  pack(F(54), wp_im, 128);   // nw, invm W1
  pack(F(58), wp_ii, 128);  pack(F(62), wp_ex, 128);   // invi, extdv W1
  pack(F(66), wp_vs, 128);                             // vscale W1
  packh(F(36), hp_i1, 3);  packh(F(40), hp_i2, 3);     // head W2 packs
  packh(F(44), hp_fs, 1);  packh(F(52), hp_cp, 3);
  packh(F(48), hp_nw, 1);  packh(F(56), hp_im, 1);
  packh(F(60), hp_ii, 1);  packh(F(64), hp_ex, 3);
  packh(F(68), hp_vs, 1);

  hipMemsetAsync(net, 0, (size_t)NN * 10 * 4, stream);

  NodeArgs na;
  na.nfeat = F(5);
  na.w1 = wp_n1; na.w2 = wp_n2;
  na.b1 = F(11); na.b2 = F(13); na.g = F(14); na.bt = F(15);
  na.wh[0] = wp_nw; na.wh[1] = wp_im; na.wh[2] = wp_ii; na.wh[3] = wp_ex; na.wh[4] = wp_vs;
  na.bh[0] = F(47); na.bh[1] = F(55); na.bh[2] = F(59); na.bh[3] = F(63); na.bh[4] = F(67);
  na.w2p[0] = hp_nw; na.w2p[1] = hp_im; na.w2p[2] = hp_ii; na.w2p[3] = hp_ex; na.w2p[4] = hp_vs;
  na.b2h[0] = F(49); na.b2h[1] = F(57); na.b2h[2] = F(61); na.b2h[3] = F(65); na.b2h[4] = F(69);
  na.lat = lat; na.wnode = wnode; na.invm = invm; na.invi = invi;
  na.extdv = extdv; na.vsc = vsc;
  node_kernel<<<NN / 32, 64, 0, stream>>>(na);

  EdgeArgs ea;
  ea.ei = ei;
  ea.pos = F(1); ea.vt = F(2); ea.vtm1 = F(3); ea.om = F(4); ea.eattr = F(6);
  ea.vs_s = F(7); ea.dmin_s = F(8); ea.dmax_s = F(9);
  ea.wnf1 = wp_nf1; ea.wnf2 = wp_nf2; ea.wed1 = wp_e1; ea.wed2 = wp_e2;
  ea.win1 = wp_in1; ea.win2 = wp_in2;
  ea.bnf1 = F(23); ea.bnf2 = F(25); ea.gnf = F(26); ea.btnf = F(27);
  ea.bed1 = F(17); ea.bed2 = F(19); ea.ged = F(20); ea.bted = F(21);
  ea.bin1 = F(29); ea.bin2 = F(31); ea.gin = F(32); ea.btin = F(33);
  ea.wh[0] = wp_i1; ea.wh[1] = wp_i2; ea.wh[2] = wp_fs; ea.wh[3] = wp_cp;
  ea.bh[0] = F(35); ea.bh[1] = F(39); ea.bh[2] = F(43); ea.bh[3] = F(51);
  ea.w2p[0] = hp_i1; ea.w2p[1] = hp_i2; ea.w2p[2] = hp_fs; ea.w2p[3] = hp_cp;
  ea.b2h[0] = F(37); ea.b2h[1] = F(41); ea.b2h[2] = F(45); ea.b2h[3] = F(53);
  ea.lat = lat; ea.wnode = wnode;
  ea.net_f = net_f; ea.net_t = net_t; ea.net_c = net_c; ea.cnt = cnt;
  edge_kernel<<<EE / 32, 64, 0, stream>>>(ea);

  finalize_kernel<<<(NN + 255) / 256, 256, 0, stream>>>(
      F(2), F(7), invm, invi, extdv, vsc, net_f, net_t, net_c, cnt, (float*)d_out);
}
